// GloCrop_UNet_60017872994915
// MI455X (gfx1250) — matrix operand order repaired and checked by reading, not compiled, not run on hardware
//
#include <hip/hip_runtime.h>
#include <hip/hip_bf16.h>

#define Bq 64
#define Nq 784
#define Cq 256
#define Hq 28
#define Wq 28
#define Mq (Bq * Nq) /* 50176 tokens */

typedef __attribute__((ext_vector_type(16))) __bf16 v16bf;
typedef __attribute__((ext_vector_type(8)))  float  v8f;

// ---------- helpers ----------
static __device__ __forceinline__ unsigned short f2bf(float f) {
  union { float f; unsigned int u; } v; v.f = f;
  unsigned int r = v.u + 0x7FFFu + ((v.u >> 16) & 1u); // round-to-nearest-even
  return (unsigned short)(r >> 16);
}
static __device__ __forceinline__ unsigned short to_bf(float f) { return f2bf(f); }
static __device__ __forceinline__ unsigned short to_bf(unsigned short h) { return h; }

// ---------- weight packing into WMMA-B register layout ----------
// Packed tile = 32(K) x 16(N) bf16, 512 elems. Within a tile, lane l holds 16
// contiguous bf16: element i = W[kt*32 + (i>>3)*16 + (l>>4)*8 + (i&7)][nt*16 + (l&15)].
// Tile order: [nt][kt] so consecutive k-steps are contiguous.
__global__ void pack_b(const float* __restrict__ W, unsigned short* __restrict__ out,
                       int K, int N, int transpose) {
  int g = blockIdx.x * blockDim.x + threadIdx.x;
  if (g >= K * N) return;
  int tile = g >> 9;
  int r    = g & 511;
  int lane = r >> 4;
  int i    = r & 15;
  int ktiles = K >> 5;
  int nt = tile / ktiles;
  int kt = tile - nt * ktiles;
  int k  = kt * 32 + ((i >> 3) << 4) + ((lane >> 4) << 3) + (i & 7);
  int n  = nt * 16 + (lane & 15);
  float w = transpose ? W[(size_t)n * K + k] : W[(size_t)k * N + n];
  out[g] = f2bf(w);
}

// fold BN (inference) into per-channel scale/shift
__global__ void bnprep(const float* __restrict__ g, const float* __restrict__ b,
                       const float* __restrict__ m, const float* __restrict__ v,
                       float* __restrict__ sc, float* __restrict__ sh) {
  int c = threadIdx.x;
  float s = g[c] * rsqrtf(v[c] + 1e-5f);
  sc[c] = s;
  sh[c] = b[c] - m[c] * s;
}

// ---------- per-channel roll (torch.roll semantics) via LDS shear ----------
// out[b,h,w,c] = in[b,(h - sign*c) mod H, w, c]   (axis 0, rows)
// out[b,h,w,c] = in[b, h,(w - sign*c) mod W, c]   (axis 1, cols)
// A block owns fixed (b, other-coordinate, 64-channel chunk); the 28x64 tile is
// loaded coalesced into LDS, then written back coalesced at sheared rows.
// LDS address = r*64 + c  ->  bank = c mod 64: conflict-free gather.
template <typename T>
static __device__ __forceinline__ void shift_body(const T* __restrict__ in,
                                                  unsigned short* __restrict__ out,
                                                  int sign, int axis) {
  __shared__ unsigned short tile[28 * 64];
  int blk = blockIdx.x;
  int cg  = blk & 3;          // which 64-channel chunk
  int bw  = blk >> 2;
  int b   = bw / 28;
  int fix = bw - b * 28;      // fixed w (axis 0) or fixed h (axis 1)
  int tid = threadIdx.x;
#pragma unroll
  for (int k = 0; k < 7; ++k) {            // 28*64 = 1792 = 7*256
    int e  = tid + k * 256;
    int r  = e >> 6;                        // source h (axis0) / w (axis1)
    int cl = e & 63;
    int n  = (axis == 0) ? (r * Wq + fix) : (fix * Wq + r);
    tile[e] = to_bf(in[((size_t)b * Nq + n) * Cq + cg * 64 + cl]);
  }
  __syncthreads();
#pragma unroll
  for (int k = 0; k < 7; ++k) {
    int e  = tid + k * 256;
    int r  = e >> 6;                        // destination h / w
    int cl = e & 63;
    int c  = cg * 64 + cl;
    int rs = ((r - sign * c) % 28 + 28) % 28;
    int n  = (axis == 0) ? (r * Wq + fix) : (fix * Wq + r);
    out[((size_t)b * Nq + n) * Cq + c] = tile[rs * 64 + cl];
  }
}
__global__ __launch_bounds__(256) void shift_f32(const float* in, unsigned short* out,
                                                 int sign, int axis) { shift_body(in, out, sign, axis); }
__global__ __launch_bounds__(256) void shift_b16(const unsigned short* in, unsigned short* out,
                                                 int sign, int axis) { shift_body(in, out, sign, axis); }

// ---------- WMMA bf16 GEMM: out[M,256] = A[M,K] @ Wp + bias (+epilogue) ----------
// 256 threads = 8 waves. Wave (0..7): rows = blk*32 + (wave>>2)*16, cols = (wave&3)*64.
// Each wave owns a 16x64 tile (4 x v8f accumulators). The k-loop is ping-pong
// buffered AND fully unrolled so every fragment is an SSA value: loads for step
// k+1 issue ahead of step k's v_wmma with no cross-iteration register copies.
template <int K, bool DO_GELU, bool DO_RESID, bool DO_RELUBN, bool DO_F32, bool DO_BF16>
__global__ __launch_bounds__(256) void gemm_wmma(
    const unsigned short* __restrict__ A, const unsigned short* __restrict__ Bp,
    const float* __restrict__ bias, const float* __restrict__ resid,
    const float* __restrict__ bnscale, const float* __restrict__ bnshift,
    float* __restrict__ outF, unsigned short* __restrict__ outB, int outBstride) {
  const int lane = threadIdx.x & 31;
  const int wave = threadIdx.x >> 5;
  const int mrow = blockIdx.x * 32 + (wave >> 2) * 16;
  const int ncol = (wave & 3) * 64;
  const int nt0  = ncol >> 4;
  constexpr int KT = K / 32; // 8 or 16 (even)

  v8f acc[4] = {};

  const int m = mrow + (lane & 15);
  const int khalf = (lane >> 4) * 8; // ISA 16-bit A layout: lane-halves own K halves
  const unsigned short* arow = A + (size_t)m * K;

  union AF { uint4 u[2]; v16bf v; };
  AF a0, a1;
  v16bf b0[4], b1[4];

  auto loadA = [&](AF& a, int kt) {
    a.u[0] = *(const uint4*)(arow + kt * 32 + khalf);
    a.u[1] = *(const uint4*)(arow + kt * 32 + 16 + khalf);
  };
  auto loadB = [&](v16bf* bb, int kt) {
#pragma unroll
    for (int f = 0; f < 4; ++f)
      bb[f] = *(const v16bf*)(Bp + ((size_t)((nt0 + f) * KT + kt) << 9) + lane * 16);
  };
  auto mmac = [&](AF& a, v16bf* bb) {
#pragma unroll
    for (int f = 0; f < 4; ++f)
      acc[f] = __builtin_amdgcn_wmma_f32_16x16x32_bf16(
          false, a.v, false, bb[f], (short)0, acc[f], false, false);
  };

  loadA(a0, 0);
  loadB(b0, 0);
#pragma unroll
  for (int kt = 0; kt < KT; kt += 2) {
    loadA(a1, kt + 1);            // prefetch odd step (KT even -> always valid)
    loadB(b1, kt + 1);
    mmac(a0, b0);
    if (kt + 2 < KT) {            // prefetch next even step
      loadA(a0, kt + 2);
      loadB(b0, kt + 2);
    }
    mmac(a1, b1);
  }

#pragma unroll
  for (int f = 0; f < 4; ++f) {
#pragma unroll
    for (int r = 0; r < 8; ++r) {
      int mm = mrow + r + ((lane >> 4) << 3); // C-layout: VGPR r -> M=r (+8 for hi lanes)
      int nn = ncol + f * 16 + (lane & 15);
      float v = acc[f][r] + bias[nn];
      if (DO_GELU)   v = 0.5f * v * (1.0f + erff(v * 0.70710678118654752f));
      if (DO_RELUBN) { v = fmaxf(v, 0.0f); v = v * bnscale[nn] + bnshift[nn]; }
      if (DO_RESID)  v += resid[(size_t)mm * Cq + nn];
      if (DO_F32)    outF[(size_t)mm * Cq + nn] = v;
      if (DO_BF16)   outB[(size_t)mm * outBstride + nn] = f2bf(v);
    }
  }
}

// ---------- LayerNorm over concat([x_1,x_2], dim=-1) (512), wave per token ----------
__global__ __launch_bounds__(256) void ln_kernel(const float* __restrict__ x1,
                                                 const float* __restrict__ x2,
                                                 const float* __restrict__ g,
                                                 const float* __restrict__ b,
                                                 unsigned short* __restrict__ out) {
  int wave = threadIdx.x >> 5, lane = threadIdx.x & 31;
  int t = blockIdx.x * 8 + wave;
  const float* src = (lane < 16) ? (x1 + (size_t)t * Cq + lane * 16)
                                 : (x2 + (size_t)t * Cq + (lane - 16) * 16);
  float vals[16];
  float s = 0.f, ss = 0.f;
#pragma unroll
  for (int i = 0; i < 16; ++i) { float v = src[i]; vals[i] = v; s += v; ss += v * v; }
#pragma unroll
  for (int off = 16; off >= 1; off >>= 1) {
    s  += __shfl_xor(s,  off, 32);
    ss += __shfl_xor(ss, off, 32);
  }
  float mean = s * (1.0f / 512.0f);
  float var  = ss * (1.0f / 512.0f) - mean * mean;
  float inv  = rsqrtf(var + 1e-5f);
  int j0 = lane * 16;
  unsigned short* dst = out + (size_t)t * 512 + j0;
#pragma unroll
  for (int i = 0; i < 16; ++i)
    dst[i] = f2bf((vals[i] - mean) * inv * g[j0 + i] + b[j0 + i]);
}

// ---------- depthwise 3x3 (+bias), token-major in/out, bf16 out ----------
__global__ __launch_bounds__(256) void dwconv(const float* __restrict__ x,
                                              const float* __restrict__ w9,
                                              const float* __restrict__ bias,
                                              unsigned short* __restrict__ out) {
  int t = blockIdx.x, c = threadIdx.x;
  int b = t / Nq, n = t - b * Nq;
  int h = n / Wq, w = n - h * Wq;
  const float* xb = x + (size_t)b * Nq * Cq + c;
  float acc = bias[c];
#pragma unroll
  for (int dy = 0; dy < 3; ++dy) {
    int hh = h + dy - 1;
    if ((unsigned)hh >= (unsigned)Hq) continue;
#pragma unroll
    for (int dx = 0; dx < 3; ++dx) {
      int wx = w + dx - 1;
      if ((unsigned)wx >= (unsigned)Wq) continue;
      acc += xb[(size_t)(hh * Wq + wx) * Cq] * w9[c * 9 + dy * 3 + dx];
    }
  }
  out[(size_t)t * Cq + c] = f2bf(acc);
}

// ---------- driver ----------
extern "C" void kernel_launch(void* const* d_in, const int* in_sizes, int n_in,
                              void* d_out, int out_size, void* d_ws, size_t ws_size,
                              hipStream_t stream) {
  (void)in_sizes; (void)n_in; (void)out_size; (void)ws_size;
  const float* x    = (const float*)d_in[0];
  const float* W1   = (const float*)d_in[1];  const float* b1 = (const float*)d_in[2];
  const float* W2   = (const float*)d_in[3];  const float* b2 = (const float*)d_in[4];
  const float* W3   = (const float*)d_in[5];  const float* b3 = (const float*)d_in[6];
  const float* W4   = (const float*)d_in[7];  const float* b4 = (const float*)d_in[8];
  const float* W5   = (const float*)d_in[9];  const float* b5 = (const float*)d_in[10];
  const float* W6   = (const float*)d_in[11]; const float* b6 = (const float*)d_in[12];
  const float* ln_g = (const float*)d_in[13]; const float* ln_b = (const float*)d_in[14];
  const float* dw_w = (const float*)d_in[15]; const float* dw_b = (const float*)d_in[16];
  const float* pw_w = (const float*)d_in[17]; const float* pw_b = (const float*)d_in[18];
  const float* bn_g = (const float*)d_in[19]; const float* bn_b = (const float*)d_in[20];
  const float* bn_m = (const float*)d_in[21]; const float* bn_v = (const float*)d_in[22];

  char* ws = (char*)d_ws;
  size_t off = 0;
  auto alloc = [&](size_t bytes) -> char* {
    char* p = ws + off;
    off = (off + bytes + 255) & ~(size_t)255;
    return p;
  };
  unsigned short* W1p = (unsigned short*)alloc(256 * 256 * 2);
  unsigned short* W2p = (unsigned short*)alloc(256 * 256 * 2);
  unsigned short* W3p = (unsigned short*)alloc(256 * 256 * 2);
  unsigned short* W4p = (unsigned short*)alloc(256 * 256 * 2);
  unsigned short* W5p = (unsigned short*)alloc(512 * 256 * 2);
  unsigned short* W6p = (unsigned short*)alloc(512 * 256 * 2);
  unsigned short* pwp = (unsigned short*)alloc(256 * 256 * 2);
  float* bnsc = (float*)alloc(256 * 4);
  float* bnsh = (float*)alloc(256 * 4);
  unsigned short* buf1  = (unsigned short*)alloc((size_t)Mq * Cq * 2);
  unsigned short* buf2  = (unsigned short*)alloc((size_t)Mq * Cq * 2);
  float*          fbuf1 = (float*)alloc((size_t)Mq * Cq * 4); // x_1 (+x)
  float*          fbuf2 = (float*)alloc((size_t)Mq * Cq * 4); // x_2 (+x)
  unsigned short* lnbuf = (unsigned short*)alloc((size_t)Mq * 512 * 2);
  unsigned short* catb  = (unsigned short*)alloc((size_t)Mq * 512 * 2);

  // weights -> WMMA-B layout; BN fold
  pack_b<<<(256 * 256 + 255) / 256, 256, 0, stream>>>(W1, W1p, 256, 256, 0);
  pack_b<<<(256 * 256 + 255) / 256, 256, 0, stream>>>(W2, W2p, 256, 256, 0);
  pack_b<<<(256 * 256 + 255) / 256, 256, 0, stream>>>(W3, W3p, 256, 256, 0);
  pack_b<<<(256 * 256 + 255) / 256, 256, 0, stream>>>(W4, W4p, 256, 256, 0);
  pack_b<<<(512 * 256 + 255) / 256, 256, 0, stream>>>(W5, W5p, 512, 256, 0);
  pack_b<<<(512 * 256 + 255) / 256, 256, 0, stream>>>(W6, W6p, 512, 256, 0);
  pack_b<<<(256 * 256 + 255) / 256, 256, 0, stream>>>(pw_w, pwp, 256, 256, 1); // B = pw^T
  bnprep<<<1, 256, 0, stream>>>(bn_g, bn_b, bn_m, bn_v, bnsc, bnsh);

  const int GB = Mq / 32;          // 1568 GEMM blocks
  const int SB = Bq * 28 * 4;      // 7168 shift blocks (b x fixed-coord x c-chunk)

  // branch 1: roll rows(+c) -> fc1 -> GELU -> roll cols(+c) -> fc2 (+x)
  shift_f32<<<SB, 256, 0, stream>>>(x, buf1, +1, 0);
  gemm_wmma<256, true, false, false, false, true><<<GB, 256, 0, stream>>>(
      buf1, W1p, b1, nullptr, nullptr, nullptr, nullptr, buf2, 256);
  shift_b16<<<SB, 256, 0, stream>>>(buf2, buf1, +1, 1);
  gemm_wmma<256, false, true, false, true, false><<<GB, 256, 0, stream>>>(
      buf1, W2p, b2, x, nullptr, nullptr, fbuf1, nullptr, 0);

  // branch 2: roll cols(-c) -> fc3 -> GELU -> roll rows(+c) -> fc4 (+x)
  shift_f32<<<SB, 256, 0, stream>>>(x, buf2, -1, 1);
  gemm_wmma<256, true, false, false, false, true><<<GB, 256, 0, stream>>>(
      buf2, W3p, b3, nullptr, nullptr, nullptr, nullptr, buf1, 256);
  shift_b16<<<SB, 256, 0, stream>>>(buf1, buf2, +1, 0);
  gemm_wmma<256, false, true, false, true, false><<<GB, 256, 0, stream>>>(
      buf2, W4p, b4, x, nullptr, nullptr, fbuf2, nullptr, 0);

  // LN(concat) -> fc5 (+x) into catbuf[:, 0:256]
  ln_kernel<<<Mq / 8, 256, 0, stream>>>(fbuf1, fbuf2, ln_g, ln_b, lnbuf);
  gemm_wmma<512, false, true, false, false, true><<<GB, 256, 0, stream>>>(
      lnbuf, W5p, b5, x, nullptr, nullptr, nullptr, catb, 512);

  // conv branch: dw3x3 -> pointwise (WMMA) + ReLU + BN into catbuf[:, 256:512]
  dwconv<<<Mq, 256, 0, stream>>>(x, dw_w, dw_b, buf1);
  gemm_wmma<256, false, false, true, false, true><<<GB, 256, 0, stream>>>(
      buf1, pwp, pw_b, nullptr, bnsc, bnsh, nullptr, catb + 256, 512);

  // final: concat @ W6 + b6 -> f32 output
  gemm_wmma<512, false, false, false, true, false><<<GB, 256, 0, stream>>>(
      catb, W6p, b6, nullptr, nullptr, nullptr, (float*)d_out, nullptr, 0);
}
